// gated_TPP_49718541418585
// MI455X (gfx1250) — matrix operand order repaired and checked by reading, not compiled, not run on hardware
//
#include <hip/hip_runtime.h>
#include <math.h>

typedef __attribute__((ext_vector_type(16))) _Float16 v16h;
typedef __attribute__((ext_vector_type(8)))  _Float16 v8h;
typedef __attribute__((ext_vector_type(4)))  _Float16 v4h;
typedef __attribute__((ext_vector_type(8)))  float    v8f;

constexpr int NB = 8;
constexpr int NL = 2048;
constexpr int ND = 256;
constexpr float INV_T = 1.0f / 200.0f;

static __device__ inline v8f wmma16(v16h a, v16h b, v8f c) {
  // D = A(16x32 f16) * B(32x16 f16) + C(16x16 f32)
  return __builtin_amdgcn_wmma_f32_16x16x32_f16(false, a, false, b, (short)0, c,
                                                false, false);
}

// ------- precompute per-row params: {1/ls^2, l, 2/s} (te = 16*emb[type]) -----
__global__ __launch_bounds__(256) void k_rowpar(
    const int* __restrict__ etype, const float* __restrict__ emb,
    const float* __restrict__ gate_w, const float* __restrict__ gate_b,
    const float* __restrict__ ker_w, const float* __restrict__ ker_b,
    float* __restrict__ rowpar) {
  int wave = threadIdx.x >> 5, lane = threadIdx.x & 31;
  int row = blockIdx.x * 8 + wave;
  if (row >= NB * NL) return;
  const float* e = emb + (size_t)etype[row] * ND;
  float dk = 0.f, d0 = 0.f, d1 = 0.f;
#pragma unroll
  for (int k = 0; k < 8; ++k) {
    int d = lane + 32 * k;
    float ev = e[d];
    dk += ev * ker_w[d];
    d0 += ev * gate_w[d];
    d1 += ev * gate_w[ND + d];
  }
#pragma unroll
  for (int off = 1; off < 32; off <<= 1) {
    dk += __shfl_xor(dk, off);
    d0 += __shfl_xor(d0, off);
    d1 += __shfl_xor(d1, off);
  }
  if (lane == 0) {
    float z = 16.f * dk + ker_b[0];
    float tb = 0.2f * z;
    float sp = (tb > 20.f) ? tb : log1pf(__expf(tb));
    float ls = sp * 5.f;  // softplus(0.2 z)/0.2
    float l = 1.f / (1.f + __expf(-(16.f * d0 + gate_b[0])));
    float s = 1.f / (1.f + __expf(-(16.f * d1 + gate_b[1])));
    rowpar[row * 4 + 0] = 1.f / (ls * ls);
    rowpar[row * 4 + 1] = l;
    rowpar[row * 4 + 2] = 2.f / s;
    rowpar[row * 4 + 3] = 0.f;
  }
}

// ------- temp_enc transposed to [B, D, L] in f16 (B-matrix friendly) ---------
__global__ __launch_bounds__(256) void k_tenc(const float* __restrict__ t,
                                              _Float16* __restrict__ tencT) {
  int idx = blockIdx.x * 256 + threadIdx.x;  // (b*ND + d)*NL + j
  int j = idx & (NL - 1);
  int d = (idx >> 11) & (ND - 1);
  int b = idx >> 19;
  float tv = t[b * NL + j];
  float inv = exp2f(-(float)d * 0.10381025f);  // 10000^(-2d/256)
  float arg = tv * inv;
  float val = ((d & 1) == 0) ? __sinf(arg) : __cosf(arg);
  tencT[idx] = (_Float16)val;
}

// ------- f32 -> f16 conversions ---------------------------------------------
__global__ __launch_bounds__(256) void k_noise16(const float* __restrict__ x,
                                                 _Float16* __restrict__ y) {
  int idx = blockIdx.x * 256 + threadIdx.x;
  y[idx] = (_Float16)x[idx];
}

__global__ __launch_bounds__(256) void k_w16(const float* __restrict__ wn,
                                             const float* __restrict__ wi,
                                             _Float16* __restrict__ wn16,
                                             _Float16* __restrict__ wi16) {
  int idx = blockIdx.x * 256 + threadIdx.x;
  if (idx < ND * ND) wn16[idx] = (_Float16)wn[idx];
  else wi16[idx - ND * ND] = (_Float16)wi[idx - ND * ND];
}

// ------- fused attention + layernorm + generator -----------------------------
// grid = NB * (NL/16), 128 threads (4 waves). Wave w owns D cols [64w, 64w+64).
// Softmax denominator is skipped entirely: LayerNorm is row-scale invariant.
__global__ __launch_bounds__(128) void k_main(
    const float* __restrict__ t, const float* __restrict__ rowpar,
    const _Float16* __restrict__ tencT, const _Float16* __restrict__ noise16,
    const _Float16* __restrict__ Wn16, const _Float16* __restrict__ Wi16,
    const float* __restrict__ gamma, const float* __restrict__ beta,
    const float* __restrict__ wout, const float* __restrict__ outb,
    float* __restrict__ out) {
  int blk = blockIdx.x;
  int b = blk >> 7;           // / (NL/16 = 128)
  int i0 = (blk & 127) << 4;  // row tile start
  int tid = threadIdx.x;
  int wave = tid >> 5, lane = tid & 31;
  int half = lane >> 4, lr = lane & 15;
  int dbase = wave * 64;
  int kb = half * 16;  // A layout: lane half -> K blocks {0..15} / {16..31}

  __shared__ float sh_ti[16], sh_ils2[16], sh_l[16], sh_is2[16];
  __shared__ float sh_mu[16], sh_rs[16];
  __shared__ float sh_red[4][16][2];
  __shared__ _Float16 sh_P[2][16][40];      // double-buffered 16x32 P tile, padded
  __shared__ _Float16 sh_hid[16][ND + 8];   // normalized hidden (A for generator)

  const float* tb = t + (size_t)b * NL;

  if (tid < 16) {
    int gi = b * NL + i0 + tid;
    sh_ti[tid] = t[gi];
    sh_ils2[tid] = rowpar[gi * 4 + 0];
    sh_l[tid] = rowpar[gi * 4 + 1];
    sh_is2[tid] = rowpar[gi * 4 + 2];
  }
  __syncthreads();

  // ---------- fused single-pass attention: hidden_un = exp(scores) @ tenc ----
  v8f acc[4] = {};  // acc[nt][k] = hidden[row = k + 8*half][dbase + nt*16 + lr]
  {
    // producer role: thread -> (row r, 4-col slice q)
    int r = tid >> 3, q = tid & 7;
    float ti = sh_ti[r], ils2 = sh_ils2[r], li = sh_l[r], is2 = sh_is2[r];
    int imax = i0 + r;
    int jend = i0 + 15;
    int buf = 0;
    for (int jb = 0; jb <= jend; jb += 32) {
      // ---- produce 4 unnormalized exp-scores (branchless causal mask) ----
      float4 tj4 = *(const float4*)(tb + jb + q * 4);
      float tjs[4] = {tj4.x, tj4.y, tj4.z, tj4.w};
      v4h pv;
#pragma unroll
      for (int cc = 0; cc < 4; ++cc) {
        int j = jb + q * 4 + cc;
        float dt = fabsf(ti - tjs[cc]);
        float kk = __expf(-dt * dt * ils2);                    // SE kernel
        float eg = __expf((dt * INV_T - li) * is2);            // exp(2u)
        float gt = 1.5f - __builtin_amdgcn_rcpf(eg + 1.f);     // 1+0.5*tanh(u)
        float pe = __expf(kk * gt);                            // unnormalized p
        pv[cc] = (_Float16)((j <= imax) ? pe : 0.f);
      }
      *(v4h*)&sh_P[buf][r][q * 4] = pv;
      __syncthreads();
      // ---- consume: A from LDS (contiguous k = 16*half + e), B from global --------
      v8h lo = *(const v8h*)&sh_P[buf][lr][kb];
      v8h hi = *(const v8h*)&sh_P[buf][lr][kb + 8];
      v16h A;
#pragma unroll
      for (int e = 0; e < 8; ++e) { A[e] = lo[e]; A[8 + e] = hi[e]; }
#pragma unroll
      for (int nt = 0; nt < 4; ++nt) {
        const _Float16* bp =
            tencT + ((size_t)(b * ND + dbase + nt * 16 + lr)) * NL + jb + half * 16;
        v16h Bv = *(const v16h*)bp;  // 16 contiguous K values: b128 x2
        acc[nt] = wmma16(A, Bv, acc[nt]);
      }
      buf ^= 1;
    }
  }

  // ---------- layernorm over D=256 (absorbs softmax 1/Z) ---------------------
  {
    float s1[8], s2[8];
#pragma unroll
    for (int k = 0; k < 8; ++k) {
      float a0 = acc[0][k], a1 = acc[1][k], a2 = acc[2][k], a3 = acc[3][k];
      s1[k] = a0 + a1 + a2 + a3;
      s2[k] = a0 * a0 + a1 * a1 + a2 * a2 + a3 * a3;
    }
#pragma unroll
    for (int off = 1; off < 16; off <<= 1) {
#pragma unroll
      for (int k = 0; k < 8; ++k) {
        s1[k] += __shfl_xor(s1[k], off);
        s2[k] += __shfl_xor(s2[k], off);
      }
    }
    if (lr == 0) {
#pragma unroll
      for (int k = 0; k < 8; ++k) {
        sh_red[wave][k + 8 * half][0] = s1[k];
        sh_red[wave][k + 8 * half][1] = s2[k];
      }
    }
  }
  __syncthreads();
  if (tid < 16) {
    float S1 = 0.f, S2 = 0.f;
#pragma unroll
    for (int w = 0; w < 4; ++w) { S1 += sh_red[w][tid][0]; S2 += sh_red[w][tid][1]; }
    float mu = S1 * (1.f / ND);
    float var = S2 * (1.f / ND) - mu * mu;
    sh_mu[tid] = mu;
    sh_rs[tid] = rsqrtf(var + 1e-6f);
  }
  __syncthreads();
#pragma unroll
  for (int nt = 0; nt < 4; ++nt) {
    int col = dbase + nt * 16 + lr;
    float ga = gamma[col], be = beta[col];
#pragma unroll
    for (int k = 0; k < 8; ++k) {
      int row = k + 8 * half;
      float v = (acc[nt][k] - sh_mu[row]) * sh_rs[row] * ga + be;
      sh_hid[row][col] = (_Float16)v;
    }
  }
  __syncthreads();

  // ---------- generator: h = relu(noise@Wn.T + hidden@Wi.T) via WMMA ---------
  v8f g[4] = {};
  {
    const _Float16* nrow = noise16 + (size_t)(b * NL + i0 + lr) * ND;
#pragma unroll
    for (int ks = 0; ks < 8; ++ks) {
      int k0 = ks * 32;
      v8h hlo = *(const v8h*)&sh_hid[lr][k0 + kb];
      v8h hhi = *(const v8h*)&sh_hid[lr][k0 + kb + 8];
      v8h nlo = *(const v8h*)(nrow + k0 + kb);
      v8h nhi = *(const v8h*)(nrow + k0 + kb + 8);
      v16h Ah, An;
#pragma unroll
      for (int e = 0; e < 8; ++e) {
        Ah[e] = hlo[e]; Ah[8 + e] = hhi[e];
        An[e] = nlo[e]; An[8 + e] = nhi[e];
      }
#pragma unroll
      for (int nt = 0; nt < 4; ++nt) {
        int col = dbase + nt * 16 + lr;
        // B[k][n] = W[n][k] -> contiguous along k in row-major W
        v16h Bi = *(const v16h*)(Wi16 + (size_t)col * ND + k0 + half * 16);
        v16h Bn = *(const v16h*)(Wn16 + (size_t)col * ND + k0 + half * 16);
        g[nt] = wmma16(Ah, Bi, g[nt]);
        g[nt] = wmma16(An, Bn, g[nt]);
      }
    }
  }
  // relu, project with gen_out_w, softplus
  {
    float po[8] = {};
#pragma unroll
    for (int nt = 0; nt < 4; ++nt) {
      float w = wout[dbase + nt * 16 + lr];
#pragma unroll
      for (int k = 0; k < 8; ++k) po[k] += fmaxf(g[nt][k], 0.f) * w;
    }
#pragma unroll
    for (int off = 1; off < 16; off <<= 1) {
#pragma unroll
      for (int k = 0; k < 8; ++k) po[k] += __shfl_xor(po[k], off);
    }
    if (lr == 0) {
#pragma unroll
      for (int k = 0; k < 8; ++k) sh_red[wave][k + 8 * half][0] = po[k];
    }
  }
  __syncthreads();
  if (tid < 16) {
    float S = outb[0];
#pragma unroll
    for (int w = 0; w < 4; ++w) S += sh_red[w][tid][0];
    out[(size_t)b * NL + i0 + tid] = (S > 20.f) ? S : log1pf(__expf(S));
  }
}

extern "C" void kernel_launch(void* const* d_in, const int* in_sizes, int n_in,
                              void* d_out, int out_size, void* d_ws, size_t ws_size,
                              hipStream_t stream) {
  (void)in_sizes; (void)n_in; (void)out_size;
  const int*   event_type  = (const int*)d_in[0];
  const float* event_time  = (const float*)d_in[1];
  const float* emb_weight  = (const float*)d_in[2];
  const float* gate_w      = (const float*)d_in[3];
  const float* gate_b      = (const float*)d_in[4];
  const float* ker_w       = (const float*)d_in[5];
  const float* ker_b       = (const float*)d_in[6];
  const float* ln_gamma    = (const float*)d_in[7];
  const float* ln_beta     = (const float*)d_in[8];
  const float* gen_input_w = (const float*)d_in[9];
  const float* gen_noise_w = (const float*)d_in[10];
  const float* gen_out_w   = (const float*)d_in[11];
  const float* gen_out_b   = (const float*)d_in[12];
  const float* noise       = (const float*)d_in[13];
  float* out = (float*)d_out;

  char* ws = (char*)d_ws;
  _Float16* tencT   = (_Float16*)(ws + 0);         // NB*ND*NL*2 = 8388608 B
  _Float16* noise16 = (_Float16*)(ws + 8388608);   // 8388608 B
  _Float16* Wn16    = (_Float16*)(ws + 16777216);  // 131072 B
  _Float16* Wi16    = (_Float16*)(ws + 16908288);  // 131072 B
  float*    rowpar  = (float*)(ws + 17039360);     // 262144 B
  if (ws_size < 17301504) return;

  k_rowpar<<<NB * NL / 8, 256, 0, stream>>>(event_type, emb_weight, gate_w, gate_b,
                                            ker_w, ker_b, rowpar);
  k_tenc<<<(NB * ND * NL) / 256, 256, 0, stream>>>(event_time, tencT);
  k_noise16<<<(NB * NL * ND) / 256, 256, 0, stream>>>(noise, noise16);
  k_w16<<<(2 * ND * ND) / 256, 256, 0, stream>>>(gen_noise_w, gen_input_w, Wn16, Wi16);
  k_main<<<NB * (NL / 16), 128, 0, stream>>>(event_time, rowpar, tencT, noise16,
                                             Wn16, Wi16, ln_gamma, ln_beta,
                                             gen_out_w, gen_out_b, out);
}
